// PAEmbedding_19018115187091
// MI455X (gfx1250) — matrix operand order repaired and checked by reading, not compiled, not run on hardware
//
#include <hip/hip_runtime.h>
#include <hip/hip_bf16.h>
#include <math.h>

// ---------------------------------------------------------------------------
// PAEmbedding for MI455X (gfx1250, wave32, WMMA).
// B=16, T=384, C=128, d=512, heads=4x128. Output (16,128,512) f32.
// Heavy GEMMs: v_wmma_f32_16x16x32_bf16 on fragment-major packed operands,
// 32x64 tile per wave, ping-pong double-buffered fragment loads (no reg movs).
// ---------------------------------------------------------------------------

typedef __attribute__((ext_vector_type(16))) __bf16 v16bf;
typedef __attribute__((ext_vector_type(8)))  float  v8f;

#define kB    16
#define kT    384
#define kC    128
#define kD    512
#define kRows 2048   // B*C

__host__ __device__ inline int anchor_of(int p) {
  return (p <= 18) ? 12 : ((p <= 36) ? 24 : ((p <= 72) ? 48 : 96));
}
// pinvT table: p = 4..192 sequentially, then p = 384 (trend) at the end.
// Entry for p is (p x old) row-major, scale sqrt(p/old) baked in.
__host__ __device__ inline long pinv_offset(int p) {
  long off = 0;
  int lim = (p > 192) ? 193 : p;
  for (int q = 4; q < lim; ++q) off += (long)anchor_of(q) * q;
  return off;
}

__device__ inline void interp_row(int j, int oldn, int p,
                                  int* i0, int* i1, float* w0, float* w1) {
  float src = ((float)j + 0.5f) * (float)oldn / (float)p - 0.5f;
  src = fminf(fmaxf(src, 0.0f), (float)(oldn - 1));
  int a = (int)floorf(src);
  if (a > oldn - 1) a = oldn - 1;
  if (a < 0) a = 0;
  int b = a + 1; if (b > oldn - 1) b = oldn - 1;
  float w = src - (float)a;
  *i0 = a; *i1 = b; *w0 = 1.0f - w; *w1 = w;
}

// ---------------------------------------------------------------------------
// K0: transpose x (B,T,C) -> xt (B*C, T)
// ---------------------------------------------------------------------------
__global__ void transpose_kernel(const float* __restrict__ x, float* __restrict__ xt) {
  int bt = blockIdx.x;            // 0..B*T-1
  int b = bt / kT, t = bt % kT;
  int c = threadIdx.x;            // 0..127
  xt[((size_t)(b * kC + c)) * kT + t] = x[((size_t)b * kT + t) * kC + c];
}

// ---------------------------------------------------------------------------
// K1: avg ACF (T,C): mean over batch of linear autocorr of per-(b,c)-centered rows
// ---------------------------------------------------------------------------
__global__ void acf_kernel(const float* __restrict__ xt, float* __restrict__ avg) {
  __shared__ float row[kT];
  __shared__ float meansh;
  int c = blockIdx.x, tau = threadIdx.x;   // block = 384 threads
  float acc = 0.0f;
  for (int b = 0; b < kB; ++b) {
    row[tau] = xt[((size_t)(b * kC + c)) * kT + tau];
    __syncthreads();
    if (tau == 0) {
      float s = 0.0f;
      for (int t = 0; t < kT; ++t) s += row[t];
      meansh = s / (float)kT;
    }
    __syncthreads();
    row[tau] -= meansh;
    __syncthreads();
    float a = 0.0f;
    for (int s2 = 0; s2 + tau < kT; ++s2) a += row[s2] * row[s2 + tau];
    acc += a;
    __syncthreads();
  }
  avg[(size_t)tau * kC + c] = acc / (float)kB;
}

// ---------------------------------------------------------------------------
// K2: peak detect + top-2 per channel (emulates jax.lax.top_k tie semantics)
// ---------------------------------------------------------------------------
__global__ void periods_kernel(const float* __restrict__ avg, int* __restrict__ periods) {
  int c = threadIdx.x;  // block = 128
  const float NEG = -3.0e38f;
  float bv1 = NEG, bv2 = NEG; int bi1 = 0, bi2 = 1;
  for (int t = 0; t < kT; ++t) {
    float cur = (t < 4) ? NEG : avg[(size_t)t * kC + c];
    int tp = (t == 0) ? (kT - 1) : (t - 1);
    float prev = (tp < 4) ? NEG : avg[(size_t)tp * kC + c];
    int tn = (t == kT - 1) ? 0 : (t + 1);
    float nxt = (tn < 4) ? NEG : avg[(size_t)tn * kC + c];
    bool peak = (cur > prev) && (cur > nxt) && (cur > 0.0f);
    float m = peak ? cur : NEG;
    if (m > bv1)      { bv2 = bv1; bi2 = bi1; bv1 = m; bi1 = t; }
    else if (m > bv2) { bv2 = m; bi2 = t; }
  }
  int p1 = bi1; if (p1 < 4) p1 = 4; if (p1 > 192) p1 = 192;
  int p2 = bi2; if (p2 < 4) p2 = 4; if (p2 > 192) p2 = 192;
  periods[c * 2 + 0] = p1;
  periods[c * 2 + 1] = p2;
}

// ---------------------------------------------------------------------------
// K3: build pinvT_p = pinv(M_p).T * sqrt(p/old) for p in {4..192, 384}.
// Full rank: p>=old -> (M^T M)^-1 M^T ; p<old -> M^T (M M^T)^-1.
// Gram matrix inverted by Gauss-Jordan in LDS (n<=96, SPD, no pivoting).
// ---------------------------------------------------------------------------
__global__ void pinv_build_kernel(float* __restrict__ pinvTable) {
  extern __shared__ float sm3[];
  const int tid = threadIdx.x, nt = blockDim.x;
  int bid = blockIdx.x;
  int p = (bid == 189) ? 384 : (4 + bid);
  int oldn = anchor_of(p);
  int n = (p >= oldn) ? oldn : p;

  float* G    = sm3;                // 96*96
  float* Inv  = sm3 + 96 * 96;      // 96*96
  float* fcol = Inv + 96 * 96;      // 96

  for (int i = tid; i < n * n; i += nt) {
    G[i] = 0.0f;
    Inv[i] = (i % (n + 1) == 0) ? 1.0f : 0.0f;
  }
  __syncthreads();
  if (tid == 0) {
    if (p >= oldn) {                // G = M^T M  (old x old)
      for (int j = 0; j < p; ++j) {
        int i0, i1; float w0, w1;
        interp_row(j, oldn, p, &i0, &i1, &w0, &w1);
        G[i0 * n + i0] += w0 * w0; G[i0 * n + i1] += w0 * w1;
        G[i1 * n + i0] += w1 * w0; G[i1 * n + i1] += w1 * w1;
      }
    } else {                        // G = M M^T  (p x p)
      for (int j = 0; j < p; ++j) {
        int ji0, ji1; float jw0, jw1;
        interp_row(j, oldn, p, &ji0, &ji1, &jw0, &jw1);
        for (int k = 0; k < p; ++k) {
          int ki0, ki1; float kw0, kw1;
          interp_row(k, oldn, p, &ki0, &ki1, &kw0, &kw1);
          float s = 0.0f;
          if (ji0 == ki0) s += jw0 * kw0;
          if (ji0 == ki1) s += jw0 * kw1;
          if (ji1 == ki0) s += jw1 * kw0;
          if (ji1 == ki1) s += jw1 * kw1;
          G[j * n + k] += s;
        }
      }
    }
  }
  __syncthreads();
  for (int col = 0; col < n; ++col) {
    float piv = G[col * n + col];
    __syncthreads();
    float ip = 1.0f / piv;
    for (int j = tid; j < n; j += nt) { G[col * n + j] *= ip; Inv[col * n + j] *= ip; }
    __syncthreads();
    for (int r = tid; r < n; r += nt) fcol[r] = (r == col) ? 0.0f : G[r * n + col];
    __syncthreads();
    for (int idx = tid; idx < n * n; idx += nt) {
      int r = idx / n, j = idx % n;
      if (r != col) {
        float f = fcol[r];
        G[idx]   -= f * G[col * n + j];
        Inv[idx] -= f * Inv[col * n + j];
      }
    }
    __syncthreads();
  }
  float scale = sqrtf((float)p / (float)oldn);
  float* out = pinvTable + pinv_offset(p);
  if (p >= oldn) {
    for (int idx = tid; idx < p * oldn; idx += nt) {
      int j = idx / oldn, o = idx % oldn;
      int i0, i1; float w0, w1;
      interp_row(j, oldn, p, &i0, &i1, &w0, &w1);
      out[idx] = scale * (Inv[o * n + i0] * w0 + Inv[o * n + i1] * w1);
    }
  } else {
    for (int idx = tid; idx < p * oldn; idx += nt) {
      int j = idx / oldn, o = idx % oldn;
      float s = 0.0f;
      for (int k = 0; k < p; ++k) {
        int i0, i1; float w0, w1;
        interp_row(k, oldn, p, &i0, &i1, &w0, &w1);
        float m = ((i0 == o) ? w0 : 0.0f) + ((i1 == o) ? w1 : 0.0f);
        s += m * Inv[k * n + j];
      }
      out[idx] = scale * s;
    }
  }
}

// ---------------------------------------------------------------------------
// Trend: combined[row, 1024:1536] = (xt_row @ pinvT_384) @ A96^T + shared_bias
// ---------------------------------------------------------------------------
__global__ void trend_kernel(const float* __restrict__ xt, const float* __restrict__ pinvTable,
                             const float* __restrict__ a96, const float* __restrict__ sbias,
                             float* __restrict__ combined) {
  extern __shared__ float smT[];
  float* rowv = smT;            // 384
  float* u    = smT + kT;       // 96
  int row = blockIdx.x, tid = threadIdx.x;
  const float* src = xt + (size_t)row * kT;
  for (int t = tid; t < kT; t += 128) rowv[t] = src[t];
  __syncthreads();
  const float* P = pinvTable + pinv_offset(384);
  for (int o = tid; o < 96; o += 128) {
    float s = 0.0f;
    for (int t = 0; t < kT; ++t) s += rowv[t] * P[t * 96 + o];
    u[o] = s;
  }
  __syncthreads();
  float* dst = combined + (size_t)row * 1536 + 1024;
  for (int d = tid; d < kD; d += 128) {
    float s = sbias[d];
    const float* w = a96 + (size_t)d * 96;
    for (int o = 0; o < 96; ++o) s += u[o] * w[o];
    dst[d] = s;
  }
}

// ---------------------------------------------------------------------------
// S1: last patch token per (b,c) for this slot -> Tlast (2048 x 512)
// ---------------------------------------------------------------------------
__global__ void tlast_kernel(const float* __restrict__ xt, const int* __restrict__ periods,
                             const float* __restrict__ pinvTable,
                             const float* __restrict__ a12, const float* __restrict__ a24,
                             const float* __restrict__ a48, const float* __restrict__ a96,
                             const float* __restrict__ sbias, float* __restrict__ Tlast, int slot) {
  extern __shared__ float smL[];
  int row = blockIdx.x, tid = threadIdx.x;
  int c = row % kC;
  int p = periods[c * 2 + slot];
  int oldn = anchor_of(p);
  int nump = kT / p;
  const float* P = pinvTable + pinv_offset(p);
  const float* A = (oldn == 12) ? a12 : ((oldn == 24) ? a24 : ((oldn == 48) ? a48 : a96));
  float* prow = smL;            // up to 192
  float* u    = smL + 192;      // up to 96
  const float* src = xt + (size_t)row * kT + (size_t)(nump - 1) * p;
  for (int t = tid; t < p; t += 128) prow[t] = src[t];
  __syncthreads();
  for (int o = tid; o < oldn; o += 128) {
    float s = 0.0f;
    for (int t = 0; t < p; ++t) s += prow[t] * P[t * oldn + o];
    u[o] = s;
  }
  __syncthreads();
  float* dst = Tlast + (size_t)row * kD;
  for (int d = tid; d < kD; d += 128) {
    float s = sbias[d];
    const float* w = A + (size_t)d * oldn;
    for (int o = 0; o < oldn; ++o) s += u[o] * w[o];
    dst[d] = s;
  }
}

// ---------------------------------------------------------------------------
// S4: per sequence — recompute tokens, scores via qk, softmax, TBar (in-place over qk)
// scores[h,j] = (qk_h . t_j + Q_h . bk_h) / sqrt(128)
// ---------------------------------------------------------------------------
__global__ void attn_core_kernel(const float* __restrict__ xt, const int* __restrict__ periods,
                                 const float* __restrict__ pinvTable,
                                 const float* __restrict__ a12, const float* __restrict__ a24,
                                 const float* __restrict__ a48, const float* __restrict__ a96,
                                 const float* __restrict__ sbias, const float* __restrict__ in_b,
                                 const float* __restrict__ Qbuf, float* __restrict__ qk_tbar,
                                 int slot) {
  extern __shared__ float smA[];
  const int nt = 128;
  int row = blockIdx.x, tid = threadIdx.x;
  int c = row % kC;
  int p = periods[c * 2 + slot];
  int oldn = anchor_of(p);
  int nump = kT / p;
  const float* P = pinvTable + pinv_offset(p);
  const float* A = (oldn == 12) ? a12 : ((oldn == 24) ? a24 : ((oldn == 48) ? a48 : a96));

  float* xrow = smA;                         // 384
  float* U    = xrow + kT;                   // nump*oldn
  float* tok  = U + nump * oldn;             // nump*512
  float* sc   = tok + (size_t)nump * kD;     // 4*nump
  float* tbar = sc + 4 * nump;               // 2048
  float* sb   = tbar + 4 * kD;               // 4

  const float* src = xt + (size_t)row * kT;
  for (int t = tid; t < kT; t += nt) xrow[t] = src[t];
  __syncthreads();
  for (int idx = tid; idx < nump * oldn; idx += nt) {
    int j = idx / oldn, o = idx % oldn;
    float s = 0.0f;
    const float* xr = xrow + j * p;
    for (int t = 0; t < p; ++t) s += xr[t] * P[t * oldn + o];
    U[idx] = s;
  }
  __syncthreads();
  for (int idx = tid; idx < nump * kD; idx += nt) {
    int j = idx / kD, d = idx % kD;
    float s = sbias[d];
    const float* w = A + (size_t)d * oldn;
    const float* uj = U + j * oldn;
    for (int o = 0; o < oldn; ++o) s += uj[o] * w[o];
    tok[idx] = s;
  }
  if (tid < 4) {
    float s = 0.0f;
    const float* q  = Qbuf + (size_t)row * kD + tid * 128;
    const float* bk = in_b + kD + tid * 128;
    for (int e = 0; e < 128; ++e) s += q[e] * bk[e];
    sb[tid] = s;
  }
  __syncthreads();
  const float inv_sqrt_hd = 0.08838834764831845f; // 1/sqrt(128)
  const float* qkrow = qk_tbar + (size_t)row * 2048;
  for (int idx = tid; idx < 4 * nump; idx += nt) {
    int h = idx / nump, j = idx % nump;
    float s = 0.0f;
    const float* qh = qkrow + h * kD;
    const float* tj = tok + (size_t)j * kD;
    for (int d = 0; d < kD; ++d) s += qh[d] * tj[d];
    sc[h * nump + j] = (s + sb[h]) * inv_sqrt_hd;
  }
  __syncthreads();
  if (tid < 4) {
    int h = tid;
    float mx = -3.0e38f;
    for (int j = 0; j < nump; ++j) mx = fmaxf(mx, sc[h * nump + j]);
    float sum = 0.0f;
    for (int j = 0; j < nump; ++j) { float e = expf(sc[h * nump + j] - mx); sc[h * nump + j] = e; sum += e; }
    float inv = 1.0f / sum;
    for (int j = 0; j < nump; ++j) sc[h * nump + j] *= inv;
  }
  __syncthreads();
  for (int idx = tid; idx < 4 * kD; idx += nt) {
    int h = idx / kD, d = idx % kD;
    float s = 0.0f;
    for (int j = 0; j < nump; ++j) s += sc[h * nump + j] * tok[(size_t)j * kD + d];
    tbar[idx] = s;
  }
  __syncthreads();
  float* dst = qk_tbar + (size_t)row * 2048;   // overwrite dead qk row with TBar
  for (int idx = tid; idx < 2048; idx += nt) dst[idx] = tbar[idx];
}

// ---------------------------------------------------------------------------
// S7: residual + LayerNorm -> combined[:, slot*512 : slot*512+512]
// ---------------------------------------------------------------------------
__global__ void ln_res_kernel(const float* __restrict__ Tlast, const float* __restrict__ Y,
                              const float* __restrict__ g, const float* __restrict__ bta,
                              float* __restrict__ combined, int slot) {
  __shared__ float rs[256], rq[256];
  int row = blockIdx.x, tid = threadIdx.x;
  float y0 = Tlast[(size_t)row * kD + tid]       + Y[(size_t)row * kD + tid];
  float y1 = Tlast[(size_t)row * kD + 256 + tid] + Y[(size_t)row * kD + 256 + tid];
  rs[tid] = y0 + y1;
  rq[tid] = y0 * y0 + y1 * y1;
  __syncthreads();
  for (int s = 128; s > 0; s >>= 1) {
    if (tid < s) { rs[tid] += rs[tid + s]; rq[tid] += rq[tid + s]; }
    __syncthreads();
  }
  float mu  = rs[0] / 512.0f;
  float var = rq[0] / 512.0f - mu * mu;
  float inv = rsqrtf(var + 1e-5f);
  float* dst = combined + (size_t)row * 1536 + slot * kD;
  dst[tid]       = (y0 - mu) * inv * g[tid]       + bta[tid];
  dst[256 + tid] = (y1 - mu) * inv * g[256 + tid] + bta[256 + tid];
}

// ---------------------------------------------------------------------------
// Fragment packers: convert f32 operands into fragment-major bf16 tiles so a
// lane's WMMA fragment is one contiguous 32B load (2x global_load_b128).
// A-tile (16x32): lane = h*16 + r; element e<8 -> K=h*8+e, e>=8 -> K=h*8+8+e.
// B-tile (32x16): same K order as A-tile; lane = h*16 + n%16.
// Packed index (as v16bf array): tile*32 + lane.
// ---------------------------------------------------------------------------
__global__ void packA_kernel(const float* __restrict__ A, int lda, int M, int K,
                             __bf16* __restrict__ out) {
  int idx = blockIdx.x * 256 + threadIdx.x;
  if (idx >= M * K) return;
  int m = idx / K, k = idx % K;
  int mt = m >> 4, kt = k >> 5;
  int r = m & 15, kk = k & 31;
  int h = (kk >> 3) & 1;
  int e = (kk & 7) + ((kk & 16) ? 8 : 0);
  int KT = K >> 5;
  size_t o = (((size_t)(mt * KT + kt)) * 32 + (h * 16 + r)) * 16 + e;
  out[o] = (__bf16)A[(size_t)m * lda + k];
}

__global__ void packB_kernel(const float* __restrict__ B, int bsk, int bsn, int K, int N,
                             __bf16* __restrict__ out) {
  int idx = blockIdx.x * 256 + threadIdx.x;
  if (idx >= K * N) return;
  int k = idx / N, n = idx % N;
  int kt = k >> 5, ntile = n >> 4;
  int lane = (((k >> 3) & 1) ? 16 : 0) + (n & 15);
  int e = (k & 7) + ((k & 16) ? 8 : 0);
  int KT = K >> 5;
  size_t o = (((size_t)(ntile * KT + kt)) * 32 + lane) * 16 + e;
  out[o] = (__bf16)B[(size_t)k * bsk + (size_t)n * bsn];
}

// ---------------------------------------------------------------------------
// bf16 WMMA GEMM on packed operands: one wave per 32x64 tile.
// Ping-pong double buffering: even K-slabs compute from (a,b) while (na,nb)
// loads, odd slabs the reverse — no register rotation movs in steady state.
// Requires KT = K/32 even (all call sites have K % 64 == 0).
// M%32==0, N%64==0. C f32 row-major, optional bias and exact GELU.
// ---------------------------------------------------------------------------
__global__ __launch_bounds__(32)
void wmma_gemm_kernel(const __bf16* __restrict__ Apk, const __bf16* __restrict__ Bpk,
                      float* __restrict__ C, int ldc, const float* __restrict__ bias,
                      int M, int N, int K, int act) {
  const int lane  = threadIdx.x & 31;
  const int r     = lane & 15;
  const int khalf = lane >> 4;
  const int mt0 = blockIdx.x * 2;   // two 16-row tiles per wave
  const int nt0 = blockIdx.y * 4;   // four 16-col subtiles per wave
  const int KT  = K >> 5;

  const v16bf* Ap = (const v16bf*)Apk;
  const v16bf* Bp = (const v16bf*)Bpk;

  v8f acc[2][4];
#pragma unroll
  for (int i = 0; i < 2; ++i)
#pragma unroll
    for (int t = 0; t < 4; ++t)
#pragma unroll
      for (int e = 0; e < 8; ++e) acc[i][t][e] = 0.0f;

  v16bf a[2], b[4], na[2], nb[4];
#pragma unroll
  for (int i = 0; i < 2; ++i)
    a[i] = Ap[((size_t)((mt0 + i) * KT)) * 32 + lane];
#pragma unroll
  for (int t = 0; t < 4; ++t)
    b[t] = Bp[((size_t)((nt0 + t) * KT)) * 32 + lane];

#pragma unroll 1
  for (int kt = 0; kt < KT; kt += 2) {
    // issue loads for slab kt+1 (always exists: KT even)
#pragma unroll
    for (int i = 0; i < 2; ++i)
      na[i] = Ap[((size_t)((mt0 + i) * KT + kt + 1)) * 32 + lane];
#pragma unroll
    for (int t = 0; t < 4; ++t)
      nb[t] = Bp[((size_t)((nt0 + t) * KT + kt + 1)) * 32 + lane];
    // compute slab kt from (a,b)
#pragma unroll
    for (int i = 0; i < 2; ++i)
#pragma unroll
      for (int t = 0; t < 4; ++t)
        acc[i][t] = __builtin_amdgcn_wmma_f32_16x16x32_bf16(
            false, a[i], false, b[t], (short)0, acc[i][t], false, false);
    // issue loads for slab kt+2 if it exists
    if (kt + 2 < KT) {
#pragma unroll
      for (int i = 0; i < 2; ++i)
        a[i] = Ap[((size_t)((mt0 + i) * KT + kt + 2)) * 32 + lane];
#pragma unroll
      for (int t = 0; t < 4; ++t)
        b[t] = Bp[((size_t)((nt0 + t) * KT + kt + 2)) * 32 + lane];
    }
    // compute slab kt+1 from (na,nb)
#pragma unroll
    for (int i = 0; i < 2; ++i)
#pragma unroll
      for (int t = 0; t < 4; ++t)
        acc[i][t] = __builtin_amdgcn_wmma_f32_16x16x32_bf16(
            false, na[i], false, nb[t], (short)0, acc[i][t], false, false);
  }
#pragma unroll
  for (int i = 0; i < 2; ++i) {
#pragma unroll
    for (int t = 0; t < 4; ++t) {
      const int n = (nt0 + t) * 16 + r;
      const float bn = bias ? bias[n] : 0.0f;
#pragma unroll
      for (int v = 0; v < 8; ++v) {
        const int m = (mt0 + i) * 16 + v + khalf * 8;
        float val = acc[i][t][v] + bn;
        if (act == 1) val = 0.5f * val * (1.0f + erff(val * 0.7071067811865475f)); // exact GELU
        C[(size_t)m * ldc + n] = val;
      }
    }
  }
}

// ---------------------------------------------------------------------------
// Host-side pack+GEMM wrapper
// ---------------------------------------------------------------------------
static inline void gemm_bf16(const float* A, int lda,
                             const float* B, int bsk, int bsn,
                             float* C, int ldc, const float* bias,
                             int M, int N, int K, int act,
                             __bf16* Apk, __bf16* Bpk, hipStream_t stream) {
  int na = M * K, nb = K * N;
  packA_kernel<<<(na + 255) / 256, 256, 0, stream>>>(A, lda, M, K, Apk);
  packB_kernel<<<(nb + 255) / 256, 256, 0, stream>>>(B, bsk, bsn, K, N, Bpk);
  wmma_gemm_kernel<<<dim3(M / 32, N / 64), 32, 0, stream>>>(Apk, Bpk, C, ldc, bias, M, N, K, act);
}

// ---------------------------------------------------------------------------
// Host orchestration
// ---------------------------------------------------------------------------
extern "C" void kernel_launch(void* const* d_in, const int* in_sizes, int n_in,
                              void* d_out, int out_size, void* d_ws, size_t ws_size,
                              hipStream_t stream) {
  (void)in_sizes; (void)n_in; (void)out_size; (void)ws_size;
  const float* x     = (const float*)d_in[0];
  const float* a12   = (const float*)d_in[1];
  const float* a24   = (const float*)d_in[2];
  const float* a48   = (const float*)d_in[3];
  const float* a96   = (const float*)d_in[4];
  const float* sbias = (const float*)d_in[5];
  const float* agg_in_w[2]  = {(const float*)d_in[6],  (const float*)d_in[12]};
  const float* agg_in_b[2]  = {(const float*)d_in[7],  (const float*)d_in[13]};
  const float* agg_out_w[2] = {(const float*)d_in[8],  (const float*)d_in[14]};
  const float* agg_out_b[2] = {(const float*)d_in[9],  (const float*)d_in[15]};
  const float* agg_ln_g[2]  = {(const float*)d_in[10], (const float*)d_in[16]};
  const float* agg_ln_b[2]  = {(const float*)d_in[11], (const float*)d_in[17]};
  const float* fw1 = (const float*)d_in[18];
  const float* fb1 = (const float*)d_in[19];
  const float* fw2 = (const float*)d_in[20];
  const float* fb2 = (const float*)d_in[21];
  float* out = (float*)d_out;

  float* ws = (float*)d_ws;
  size_t off = 0;
  float* xt  = ws + off; off += (size_t)kB * kC * kT;      // 786432
  float* avg = ws + off; off += (size_t)kT * kC;           // 49152
  int*   periods = (int*)(ws + off); off += 256;
  long ptot = pinv_offset(384) + 96l * 384;                // pinvT table total
  float* pinvT = ws + off; off += (size_t)ptot;
  off = (off + 7) & ~(size_t)7;                            // 32B align
  float* Tlast = ws + off; off += (size_t)kRows * kD;
  float* Qb    = ws + off; off += (size_t)kRows * kD;
  float* QK    = ws + off; off += (size_t)kRows * 2048;    // also holds TBar in-place
  float* Ob    = ws + off; off += (size_t)kRows * kD;
  float* Yb    = ws + off; off += (size_t)kRows * kD;
  float* comb  = ws + off; off += (size_t)kRows * 1536;
  float* Hb    = ws + off; off += (size_t)kRows * kD;
  off = (off + 7) & ~(size_t)7;
  __bf16* Apk = (__bf16*)(ws + off); off += (size_t)kRows * 1536 / 2;  // max M*K bf16
  __bf16* Bpk = (__bf16*)(ws + off); off += (size_t)1536 * kD / 2;     // max K*N bf16

  // ---- periods pipeline ----
  transpose_kernel<<<kB * kT, kC, 0, stream>>>(x, xt);
  acf_kernel<<<kC, kT, 0, stream>>>(xt, avg);
  periods_kernel<<<1, kC, 0, stream>>>(avg, periods);
  pinv_build_kernel<<<190, 256, (size_t)(2 * 96 * 96 + 96) * sizeof(float), stream>>>(pinvT);

  // ---- trend branch -> combined[:,1024:1536] ----
  trend_kernel<<<kRows, 128, (size_t)(kT + 96) * sizeof(float), stream>>>(xt, pinvT, a96, sbias, comb);

  // ---- two attention slots ----
  for (int s = 0; s < 2; ++s) {
    tlast_kernel<<<kRows, 128, (size_t)(192 + 96) * sizeof(float), stream>>>(
        xt, periods, pinvT, a12, a24, a48, a96, sbias, Tlast, s);
    // Q = Tlast @ Wq^T + bq        (B[k,n] = in_w[n*512+k] -> bsk=1, bsn=512)
    gemm_bf16(Tlast, kD, agg_in_w[s], 1, kD, Qb, kD, agg_in_b[s],
              kRows, kD, kD, 0, Apk, Bpk, stream);
    // qk_h = Q_h @ Wk_h            (B[k,n] = in_w[(512+h*128+k)*512+n] -> bsk=512, bsn=1)
    for (int h = 0; h < 4; ++h) {
      gemm_bf16(Qb + h * 128, kD,
                agg_in_w[s] + (size_t)(kD + h * 128) * kD, kD, 1,
                QK + h * kD, 2048, (const float*)nullptr,
                kRows, kD, 128, 0, Apk, Bpk, stream);
    }
    // softmax core: reads qk rows, writes TBar in-place
    attn_core_kernel<<<kRows, 128, (size_t)53200 * sizeof(float), stream>>>(
        xt, periods, pinvT, a12, a24, a48, a96, sbias, agg_in_b[s], Qb, QK, s);
    // o_h = TBar_h @ Wv_h^T + bv_h (B[k,n] = in_w[(1024+h*128+n)*512+k] -> bsk=1, bsn=512)
    for (int h = 0; h < 4; ++h) {
      gemm_bf16(QK + h * kD, 2048,
                agg_in_w[s] + (size_t)(2 * kD + h * 128) * kD, 1, kD,
                Ob + h * 128, kD, agg_in_b[s] + 2 * kD + h * 128,
                kRows, 128, kD, 0, Apk, Bpk, stream);
    }
    // Y = O @ out_w^T + out_b
    gemm_bf16(Ob, kD, agg_out_w[s], 1, kD, Yb, kD, agg_out_b[s],
              kRows, kD, kD, 0, Apk, Bpk, stream);
    // residual + LN -> combined[:, s*512 : s*512+512]
    ln_res_kernel<<<kRows, 256, 0, stream>>>(Tlast, Yb, agg_ln_g[s], agg_ln_b[s], comb, s);
  }

  // ---- fusion MLP ----
  gemm_bf16(comb, 1536, fw1, 1, 1536, Hb, kD, fb1,
            kRows, kD, 1536, 1, Apk, Bpk, stream);   // exact GELU
  gemm_bf16(Hb, kD, fw2, 1, kD, out, kD, fb2,
            kRows, kD, kD, 0, Apk, Bpk, stream);
}
